// LeNet5_T_53721450938983
// MI455X (gfx1250) — matrix operand order repaired and checked by reading, not compiled, not run on hardware
//
#include <hip/hip_runtime.h>
#include <hip/hip_bf16.h>

typedef __attribute__((ext_vector_type(16))) _Float16 v16h;
typedef __attribute__((ext_vector_type(8)))  _Float16 v8h;
typedef __attribute__((ext_vector_type(8)))  float    v8f;
typedef __attribute__((ext_vector_type(4)))  unsigned int v4u;
typedef __attribute__((ext_vector_type(8)))  int      v8i;
typedef __attribute__((ext_vector_type(4)))  int      v4i;

// ---------------------------------------------------------------------------
// WMMA helpers (CDNA5 wave32, V_WMMA_F32_16X16X32_F16)
// ---------------------------------------------------------------------------
__device__ __forceinline__ v8f wmma_f16(v16h a, v16h b, v8f c) {
  return __builtin_amdgcn_wmma_f32_16x16x32_f16(false, a, false, b, (short)0, c,
                                                false, false);
}

__device__ __forceinline__ v16h cat16(v8h lo, v8h hi) {
  return __builtin_shufflevector(lo, hi, 0, 1, 2, 3, 4, 5, 6, 7, 8, 9, 10, 11,
                                 12, 13, 14, 15);
}

// A fragment: 16x32 f16; rowptr = this lane's row (32 halves, 64B aligned).
// halves 0..7 = K kb..kb+7, halves 8..15 = K 16+kb..16+kb+7, kb = (lane>=16)*8.
// Two ds_load_b128 per lane.
__device__ __forceinline__ v16h a_frag(const _Float16* rowptr) {
  int kb = ((threadIdx.x & 31) >> 4) * 8;
  v8h lo = *(const v8h*)(rowptr + kb);
  v8h hi = *(const v8h*)(rowptr + 16 + kb);
  return cat16(lo, hi);
}

// B fragment from an N-major tile: colptr = this lane's column's 32 contiguous
// K halves.  halves 0..7 = K kb..kb+7, halves 8..15 = K 16+kb..16+kb+7, kb = (lane>=16)*8.
__device__ __forceinline__ v16h b_frag(const _Float16* colptr) {
  int koff = ((threadIdx.x & 31) >> 4) * 8;
  v8h lo = *(const v8h*)(colptr + koff);
  v8h hi = *(const v8h*)(colptr + 16 + koff);
  return cat16(lo, hi);
}

// ---------------------------------------------------------------------------
// Tensor Data Mover: 2-D tile (rows of tile_d0 halves, tile_d1 rows, row
// stride stride_d0 halves) from global -> LDS (compacted).  D# per ISA §8.3/8.4.
// data_size=1 (2-byte elements), type=2, count=1, no padding/iterate/gather.
// Issue from ONE wave; completion via s_wait_tensorcnt.
// ---------------------------------------------------------------------------
__device__ __forceinline__ void tdm_load_2d(void* lds_dst, const void* gsrc,
                                            unsigned tile_d0, unsigned tile_d1,
                                            unsigned long long stride_d0) {
  unsigned long long ga = (unsigned long long)gsrc;
  unsigned lds_off = (unsigned)(unsigned long long)lds_dst;  // LDS byte address
  v4u g0;
  g0[0] = 1u;                                   // count=1, user descriptor
  g0[1] = lds_off;                              // D#.lds_addr
  g0[2] = (unsigned)ga;                         // global_addr[31:0]
  g0[3] = (unsigned)((ga >> 32) & 0x01FFFFFFull) | 0x80000000u;  // [56:32]|type=2
  v8i g1;
  g1[0] = 0x00010000;                           // wg_mask=0, data_size=1 (2B)
  g1[1] = (int)((tile_d0 & 0xFFFFu) << 16);     // tensor_dim0[15:0]
  g1[2] = (int)(((tile_d0 >> 16) & 0xFFFFu) |
                ((tile_d1 & 0xFFFFu) << 16));   // tensor_dim0 hi | tensor_dim1 lo
  g1[3] = (int)(((tile_d1 >> 16) & 0xFFFFu) |
                ((tile_d0 & 0xFFFFu) << 16));   // tensor_dim1 hi | tile_dim0
  g1[4] = (int)(tile_d1 & 0xFFFFu);             // tile_dim1 (tile_dim2=0)
  g1[5] = (int)(stride_d0 & 0xFFFFFFFFull);     // tensor_dim0_stride lo
  g1[6] = (int)((stride_d0 >> 32) & 0xFFFFull); // stride hi, dim1_stride=0
  g1[7] = 0;
  v4i z4 = {};
#if __has_include(<hip/amd_detail/amd_gfx1250_TDM.h>)
  v8i z8 = {};
  __builtin_amdgcn_tensor_load_to_lds(g0, g1, z4, z4, z8, 0);
#else
  __builtin_amdgcn_tensor_load_to_lds(g0, g1, z4, z4, 0);
#endif
}

// ---------------------------------------------------------------------------
// Ternarize: one wave (32 lanes) per output channel.  tq[o][e] = alpha*tern.
// ---------------------------------------------------------------------------
__global__ __launch_bounds__(32) void k_ternarize(const float* __restrict__ w,
                                                  float* __restrict__ tq, int E) {
  int o = blockIdx.x;
  int lane = threadIdx.x;
  const float* wo = w + (size_t)o * E;
  float s = 0.f;
  for (int e = lane; e < E; e += 32) s += fabsf(wo[e]);
#pragma unroll
  for (int off = 16; off; off >>= 1) s += __shfl_xor(s, off, 32);
  float delta = 0.7f * s / (float)E;
  float ssum = 0.f, cnt = 0.f;
  for (int e = lane; e < E; e += 32) {
    float a = fabsf(wo[e]);
    if (a > delta) { ssum += a; cnt += 1.f; }
  }
#pragma unroll
  for (int off = 16; off; off >>= 1) {
    ssum += __shfl_xor(ssum, off, 32);
    cnt  += __shfl_xor(cnt, off, 32);
  }
  float alpha = ssum / cnt;
  for (int e = lane; e < E; e += 32) {
    float v = wo[e];
    float t = (v > delta) ? alpha : ((v < -delta) ? -alpha : 0.f);
    tq[(size_t)o * E + e] = t;
  }
}

// ---------------------------------------------------------------------------
// Weight packs: all B matrices stored N-major ([N][K]) in f16 so each lane's
// B-fragment is 32 contiguous halves in LDS.
// ---------------------------------------------------------------------------
__global__ void k_pack_conv1(const float* __restrict__ tq /*[32][25]*/,
                             _Float16* __restrict__ dst /*[32 N][32 Kpad]*/) {
  int i = blockIdx.x * blockDim.x + threadIdx.x;
  if (i >= 32 * 32) return;
  int n = i >> 5, k = i & 31;
  dst[i] = (_Float16)((k < 25) ? tq[n * 25 + k] : 0.f);
}

__global__ void k_pack_conv2(const float* __restrict__ tq /*[64][800]*/,
                             _Float16* __restrict__ dst /*[25][64 N][32 K]*/) {
  int i = blockIdx.x * blockDim.x + threadIdx.x;
  if (i >= 25 * 64 * 32) return;
  int ci = i & 31;
  int rest = i >> 5;
  int n = rest & 63;
  int t = rest >> 6;
  dst[i] = (_Float16)tq[(size_t)n * 800 + ci * 25 + t];
}

__global__ void k_pack_fc1(const float* __restrict__ tq /*[512][1024]*/,
                           _Float16* __restrict__ dst /*[512 N][1024 K]*/) {
  int i = blockIdx.x * blockDim.x + threadIdx.x;
  if (i >= 512 * 1024) return;
  int k = i & 1023;                       // NHWC flat index (y*4+x)*64 + c
  int n = i >> 10;
  int kref = (k & 63) * 16 + (k >> 6);    // NCHW flat index c*16 + (y*4+x)
  dst[i] = (_Float16)tq[(size_t)n * 1024 + kref];
}

__global__ void k_pack_fc2(const float* __restrict__ tq /*[10][512]*/,
                           _Float16* __restrict__ dst /*[16 N][512 K]*/) {
  int i = blockIdx.x * blockDim.x + threadIdx.x;
  if (i >= 16 * 512) return;
  int k = i & 511, n = i >> 9;
  dst[i] = (_Float16)((n < 10) ? tq[n * 512 + k] : 0.f);
}

// ---------------------------------------------------------------------------
// conv1: x[B,1,28,28] f32 -> h1[B*576, 32] f16 (NHWC).  Implicit GEMM K=25->32.
// Weights via TDM (contiguous 2KB); im2col A built by VALU.
// ---------------------------------------------------------------------------
__global__ __launch_bounds__(128) void k_conv1(const float* __restrict__ x,
                                               const _Float16* __restrict__ wt,
                                               const float* __restrict__ bias,
                                               _Float16* __restrict__ out) {
  __shared__ __align__(16) _Float16 As[64 * 32];
  __shared__ __align__(16) _Float16 Ws[32 * 32];  // N-major [32][32]
  int tid = threadIdx.x;
  int wv = tid >> 5, lane = tid & 31, lc = lane & 15, hi = lane >> 4;
  unsigned p0 = blockIdx.x * 64u;

  if (wv == 0) tdm_load_2d(Ws, wt, 32 * 32, 1, 32 * 32);

  for (int i = tid; i < 64 * 32; i += 128) {
    int pix = i >> 5, k = i & 31;
    unsigned p = p0 + (unsigned)pix;
    unsigned b = p / 576u, r = p % 576u;
    unsigned oy = r / 24u, ox = r % 24u;
    float v = 0.f;
    if (k < 25) {
      int dy = k / 5, dx = k % 5;
      v = x[(size_t)b * 784 + (oy + dy) * 28 + (ox + dx)];
    }
    As[i] = (_Float16)v;
  }
  if (wv == 0) __builtin_amdgcn_s_wait_tensorcnt(0);
  __syncthreads();

  v16h a = a_frag(As + (wv * 16 + lc) * 32);
  v8f c0 = {}, c1 = {};
  c0 = wmma_f16(a, b_frag(Ws + lc * 32), c0);
  c1 = wmma_f16(a, b_frag(Ws + (16 + lc) * 32), c1);

#pragma unroll
  for (int r = 0; r < 8; ++r) {
    size_t pix = (size_t)p0 + wv * 16 + r + hi * 8;
    out[pix * 32 + lc]      = (_Float16)(c0[r] + bias[lc]);
    out[pix * 32 + 16 + lc] = (_Float16)(c1[r] + bias[16 + lc]);
  }
}

// ---------------------------------------------------------------------------
// Deterministic BN statistics + finalize + fused bn/pool/relu (NHWC f16)
// ---------------------------------------------------------------------------
__global__ __launch_bounds__(256) void k_bnstats(const _Float16* __restrict__ h,
                                                 float* __restrict__ psum,
                                                 float* __restrict__ psq,
                                                 int Npix, int C, int S) {
  int c = blockIdx.x, s = blockIdx.y, tid = threadIdx.x;
  long long per = (Npix + S - 1) / S;
  long long beg = (long long)s * per;
  long long end = beg + per;
  if (end > Npix) end = Npix;
  float sum = 0.f, sq = 0.f;
  for (long long p = beg + tid; p < end; p += 256) {
    float v = (float)h[(size_t)p * C + c];
    sum += v;
    sq += v * v;
  }
  __shared__ float ls[256], lq[256];
  ls[tid] = sum; lq[tid] = sq;
  __syncthreads();
  for (int o = 128; o; o >>= 1) {
    if (tid < o) { ls[tid] += ls[tid + o]; lq[tid] += lq[tid + o]; }
    __syncthreads();
  }
  if (tid == 0) { psum[c * S + s] = ls[0]; psq[c * S + s] = lq[0]; }
}

__global__ void k_bnfinalize(const float* __restrict__ psum,
                             const float* __restrict__ psq,
                             const float* __restrict__ gamma,
                             const float* __restrict__ beta,
                             float2* __restrict__ bnp, int C, int S, float invN) {
  int c = threadIdx.x;
  if (c >= C) return;
  float s = 0.f, q = 0.f;
  for (int i = 0; i < S; ++i) { s += psum[c * S + i]; q += psq[c * S + i]; }
  float mean = s * invN;
  float var = q * invN - mean * mean;
  float sc = gamma[c] * rsqrtf(var + 1e-5f);
  bnp[c] = make_float2(sc, beta[c] - mean * sc);
}

__global__ __launch_bounds__(256) void k_bnpoolrelu(const _Float16* __restrict__ in,
                                                    _Float16* __restrict__ out,
                                                    const float2* __restrict__ bnp,
                                                    int B, int H, int W, int C) {
  int i = blockIdx.x * blockDim.x + threadIdx.x;
  int total = B * (H / 2) * (W / 2) * C;
  if (i >= total) return;
  int c = i % C;
  int t = i / C;
  int ox = t % (W / 2); t /= (W / 2);
  int oy = t % (H / 2);
  int b = t / (H / 2);
  size_t base = (((size_t)b * H + oy * 2) * W + ox * 2) * C + c;
  float2 p = bnp[c];
  float v00 = fmaf((float)in[base], p.x, p.y);
  float v01 = fmaf((float)in[base + C], p.x, p.y);
  float v10 = fmaf((float)in[base + (size_t)W * C], p.x, p.y);
  float v11 = fmaf((float)in[base + (size_t)W * C + C], p.x, p.y);
  float m = fmaxf(fmaxf(v00, v01), fmaxf(v10, v11));
  out[i] = (_Float16)fmaxf(m, 0.f);
}

// ---------------------------------------------------------------------------
// conv2: h1p[B,12,12,32] f16 -> h2[B*64, 64] f16.  One image per block.
// TDM double-buffered pipeline: A = 8 rows x 512B (stride 768B) window,
// W = contiguous 4KB tap tile ([64 N][32 K]).  25 taps x 4 WMMA per wave.
// ---------------------------------------------------------------------------
__global__ __launch_bounds__(128) void k_conv2(const _Float16* __restrict__ h1p,
                                               const _Float16* __restrict__ wt,
                                               const float* __restrict__ bias,
                                               _Float16* __restrict__ out) {
  __shared__ __align__(16) _Float16 As[2][2048];
  __shared__ __align__(16) _Float16 Ws[2][2048];
  int tid = threadIdx.x;
  int b = blockIdx.x;
  int wv = tid >> 5, lane = tid & 31, lc = lane & 15, hi = lane >> 4;
  v8f acc[4] = {};

  if (wv == 0) {
    // tap 0: dy=0, dx=0
    tdm_load_2d(&As[0][0], h1p + (size_t)(b * 12) * 12 * 32, 256, 8, 384);
    tdm_load_2d(&Ws[0][0], wt, 2048, 1, 2048);
  }

  for (int t = 0; t < 25; ++t) {
    int cur = t & 1;
    __syncthreads();  // everyone done reading buffer cur^1 from iteration t-1
    if (wv == 0) {
      if (t + 1 < 25) {
        int dy = (t + 1) / 5, dx = (t + 1) % 5;
        tdm_load_2d(&As[cur ^ 1][0],
                    h1p + ((size_t)(b * 12 + dy) * 12 + dx) * 32, 256, 8, 384);
        tdm_load_2d(&Ws[cur ^ 1][0], wt + (size_t)(t + 1) * 2048, 2048, 1, 2048);
        __builtin_amdgcn_s_wait_tensorcnt(2);  // this tap's 2 loads done
      } else {
        __builtin_amdgcn_s_wait_tensorcnt(0);
      }
    }
    __syncthreads();

    v16h a = a_frag(&As[cur][(wv * 16 + lc) * 32]);
#pragma unroll
    for (int nt = 0; nt < 4; ++nt)
      acc[nt] = wmma_f16(a, b_frag(&Ws[cur][(nt * 16 + lc) * 32]), acc[nt]);
  }

#pragma unroll
  for (int nt = 0; nt < 4; ++nt)
#pragma unroll
    for (int r = 0; r < 8; ++r) {
      size_t pix = (size_t)b * 64 + wv * 16 + r + hi * 8;
      out[pix * 64 + nt * 16 + lc] = (_Float16)(acc[nt][r] + bias[nt * 16 + lc]);
    }
}

// ---------------------------------------------------------------------------
// fc1: [4096,1024] f16 x W[512 N][1024 K] f16 -> relu -> [4096,512] f16.
// 64x64 tile, K chunks of 32, TDM double-buffered A and B tiles.
// ---------------------------------------------------------------------------
__global__ __launch_bounds__(128) void k_fc1(const _Float16* __restrict__ A,
                                             const _Float16* __restrict__ W,
                                             const float* __restrict__ bias,
                                             _Float16* __restrict__ out) {
  __shared__ __align__(16) _Float16 As[2][2048];
  __shared__ __align__(16) _Float16 Bs[2][2048];
  int tid = threadIdx.x;
  int m0 = blockIdx.x * 64;
  int n0 = blockIdx.y * 64;
  int wv = tid >> 5, lane = tid & 31, lc = lane & 15, hi = lane >> 4;
  v8f acc[4] = {};

  if (wv == 0) {
    tdm_load_2d(&As[0][0], A + (size_t)m0 * 1024, 32, 64, 1024);
    tdm_load_2d(&Bs[0][0], W + (size_t)n0 * 1024, 32, 64, 1024);
  }

  for (int it = 0; it < 32; ++it) {
    int cur = it & 1;
    __syncthreads();
    if (wv == 0) {
      if (it + 1 < 32) {
        int k1 = (it + 1) * 32;
        tdm_load_2d(&As[cur ^ 1][0], A + (size_t)m0 * 1024 + k1, 32, 64, 1024);
        tdm_load_2d(&Bs[cur ^ 1][0], W + (size_t)n0 * 1024 + k1, 32, 64, 1024);
        __builtin_amdgcn_s_wait_tensorcnt(2);  // current chunk's loads done
      } else {
        __builtin_amdgcn_s_wait_tensorcnt(0);
      }
    }
    __syncthreads();

    v16h a = a_frag(&As[cur][(wv * 16 + lc) * 32]);
#pragma unroll
    for (int nt = 0; nt < 4; ++nt)
      acc[nt] = wmma_f16(a, b_frag(&Bs[cur][(nt * 16 + lc) * 32]), acc[nt]);
  }

#pragma unroll
  for (int nt = 0; nt < 4; ++nt)
#pragma unroll
    for (int r = 0; r < 8; ++r) {
      int row = m0 + wv * 16 + r + hi * 8;
      int col = n0 + nt * 16 + lc;
      float v = acc[nt][r] + bias[col];
      out[(size_t)row * 512 + col] = (_Float16)fmaxf(v, 0.f);
    }
}

// ---------------------------------------------------------------------------
// fc2: [4096,512] f16 x W[16 N][512 K] f16 + bias -> [4096,10] f32.
// Whole W (16KB) via one TDM; A chunks double-buffered via TDM.
// ---------------------------------------------------------------------------
__global__ __launch_bounds__(128) void k_fc2(const _Float16* __restrict__ A,
                                             const _Float16* __restrict__ W,
                                             const float* __restrict__ bias,
                                             float* __restrict__ out) {
  __shared__ __align__(16) _Float16 As[2][2048];
  __shared__ __align__(16) _Float16 Wl[16 * 512];
  int tid = threadIdx.x;
  int m0 = blockIdx.x * 64;
  int wv = tid >> 5, lane = tid & 31, lc = lane & 15, hi = lane >> 4;

  if (wv == 0) {
    tdm_load_2d(Wl, W, 8192, 1, 8192);
    tdm_load_2d(&As[0][0], A + (size_t)m0 * 512, 32, 64, 512);
  }

  v8f acc = {};
  for (int it = 0; it < 16; ++it) {
    int cur = it & 1;
    int k0 = it * 32;
    __syncthreads();
    if (wv == 0) {
      if (it + 1 < 16) {
        tdm_load_2d(&As[cur ^ 1][0], A + (size_t)m0 * 512 + (it + 1) * 32,
                    32, 64, 512);
        __builtin_amdgcn_s_wait_tensorcnt(1);  // only the new load outstanding
      } else {
        __builtin_amdgcn_s_wait_tensorcnt(0);
      }
    }
    __syncthreads();

    v16h a = a_frag(&As[cur][(wv * 16 + lc) * 32]);
    acc = wmma_f16(a, b_frag(&Wl[lc * 512 + k0]), acc);
  }

  if (lc < 10) {
#pragma unroll
    for (int r = 0; r < 8; ++r) {
      int row = m0 + wv * 16 + r + hi * 8;
      out[(size_t)row * 10 + lc] = acc[r] + bias[lc];
    }
  }
}

// ---------------------------------------------------------------------------
// Host orchestration
// ---------------------------------------------------------------------------
extern "C" void kernel_launch(void* const* d_in, const int* in_sizes, int n_in,
                              void* d_out, int out_size, void* d_ws, size_t ws_size,
                              hipStream_t stream) {
  const float* x       = (const float*)d_in[0];
  const float* w_conv1 = (const float*)d_in[1];
  const float* b_conv1 = (const float*)d_in[2];
  const float* gamma1  = (const float*)d_in[3];
  const float* beta1   = (const float*)d_in[4];
  const float* w_conv2 = (const float*)d_in[5];
  const float* b_conv2 = (const float*)d_in[6];
  const float* gamma2  = (const float*)d_in[7];
  const float* beta2   = (const float*)d_in[8];
  const float* w_fc1   = (const float*)d_in[9];
  const float* b_fc1   = (const float*)d_in[10];
  const float* w_fc2   = (const float*)d_in[11];
  const float* b_fc2   = (const float*)d_in[12];
  float* outp = (float*)d_out;

  const int B = 4096;
  const long long M1 = (long long)B * 576;
  const long long M2 = (long long)B * 64;

  char* ws = (char*)d_ws;
  size_t off = 0;
  auto alloc = [&](size_t bytes) -> void* {
    void* p = ws + off;
    off = (off + bytes + 255) & ~(size_t)255;
    return p;
  };

  float*    tq1  = (float*)alloc(32 * 25 * 4);
  float*    tq2  = (float*)alloc(64 * 800 * 4);
  float*    tq3  = (float*)alloc((size_t)512 * 1024 * 4);
  float*    tq4  = (float*)alloc(10 * 512 * 4);
  _Float16* w1t  = (_Float16*)alloc(32 * 32 * 2);
  _Float16* w2t  = (_Float16*)alloc(25 * 64 * 32 * 2);
  _Float16* wf1  = (_Float16*)alloc((size_t)512 * 1024 * 2);
  _Float16* wf2  = (_Float16*)alloc(16 * 512 * 2);
  _Float16* h1   = (_Float16*)alloc((size_t)M1 * 32 * 2);
  _Float16* h1p  = (_Float16*)alloc((size_t)B * 144 * 32 * 2);
  _Float16* h2   = (_Float16*)alloc((size_t)M2 * 64 * 2);
  _Float16* h2p  = (_Float16*)alloc((size_t)B * 16 * 64 * 2);
  _Float16* hf1  = (_Float16*)alloc((size_t)B * 512 * 2);
  float*    ps1  = (float*)alloc(32 * 64 * 4);
  float*    pq1  = (float*)alloc(32 * 64 * 4);
  float*    ps2  = (float*)alloc(64 * 64 * 4);
  float*    pq2  = (float*)alloc(64 * 64 * 4);
  float2*   bnp1 = (float2*)alloc(32 * sizeof(float2));
  float2*   bnp2 = (float2*)alloc(64 * sizeof(float2));

  k_ternarize<<<32, 32, 0, stream>>>(w_conv1, tq1, 25);
  k_ternarize<<<64, 32, 0, stream>>>(w_conv2, tq2, 800);
  k_ternarize<<<512, 32, 0, stream>>>(w_fc1, tq3, 1024);
  k_ternarize<<<10, 32, 0, stream>>>(w_fc2, tq4, 512);

  k_pack_conv1<<<(32 * 32 + 255) / 256, 256, 0, stream>>>(tq1, w1t);
  k_pack_conv2<<<(25 * 64 * 32 + 255) / 256, 256, 0, stream>>>(tq2, w2t);
  k_pack_fc1<<<(512 * 1024 + 255) / 256, 256, 0, stream>>>(tq3, wf1);
  k_pack_fc2<<<(16 * 512 + 255) / 256, 256, 0, stream>>>(tq4, wf2);

  k_conv1<<<(unsigned)(M1 / 64), 128, 0, stream>>>(x, w1t, b_conv1, h1);

  k_bnstats<<<dim3(32, 64), 256, 0, stream>>>(h1, ps1, pq1, (int)M1, 32, 64);
  k_bnfinalize<<<1, 64, 0, stream>>>(ps1, pq1, gamma1, beta1, bnp1, 32, 64,
                                     1.0f / (float)M1);
  k_bnpoolrelu<<<(B * 144 * 32 + 255) / 256, 256, 0, stream>>>(h1, h1p, bnp1,
                                                               B, 24, 24, 32);

  k_conv2<<<B, 128, 0, stream>>>(h1p, w2t, b_conv2, h2);

  k_bnstats<<<dim3(64, 64), 256, 0, stream>>>(h2, ps2, pq2, (int)M2, 64, 64);
  k_bnfinalize<<<1, 64, 0, stream>>>(ps2, pq2, gamma2, beta2, bnp2, 64, 64,
                                     1.0f / (float)M2);
  k_bnpoolrelu<<<(B * 16 * 64 + 255) / 256, 256, 0, stream>>>(h2, h2p, bnp2,
                                                              B, 8, 8, 64);

  k_fc1<<<dim3(B / 64, 512 / 64), 128, 0, stream>>>(h2p, wf1, b_fc1, hf1);
  k_fc2<<<B / 64, 128, 0, stream>>>(hf1, wf2, b_fc2, outp);
}
